// Decoder_35837207118458
// MI455X (gfx1250) — matrix operand order repaired and checked by reading, not compiled, not run on hardware
//
#include <hip/hip_runtime.h>
#include <hip/hip_bf16.h>

// ---- problem dims ----
constexpr int EDIM  = 512;
constexpr int HDIM  = 1024;
constexpr int SLEN  = 256;
constexpr int BATCH = 128;
constexpr int CHDIM = 2048;   // 2*H

typedef __attribute__((ext_vector_type(16))) __bf16 v16bf;
typedef __attribute__((ext_vector_type(8)))  float  v8f;
typedef __attribute__((ext_vector_type(4)))  int    v4i;

union FragU { v16bf v; unsigned u[8]; uint4 q[2]; };

#if defined(__gfx1250__) && \
    __has_builtin(__builtin_amdgcn_global_load_async_to_lds_b128) && \
    __has_builtin(__builtin_amdgcn_s_wait_asynccnt)
#define USE_ASYNC_LDS 1
#else
#define USE_ASYNC_LDS 0
#endif

#define AS1 __attribute__((address_space(1)))
#define AS3 __attribute__((address_space(3)))

// pack two f32 into one dword holding two bf16 (RNE-ish rounding)
__device__ __forceinline__ unsigned pk2(float a, float b) {
    unsigned ua = __builtin_bit_cast(unsigned, a);
    unsigned ub = __builtin_bit_cast(unsigned, b);
    ua = (ua + 0x7FFFu + ((ua >> 16) & 1u)) >> 16;
    ub = (ub + 0x7FFFu + ((ub >> 16) & 1u)) & 0xFFFF0000u;
    return ua | ub;
}

__device__ __forceinline__ float sigmoidf_(float x) {
    return 1.0f / (1.0f + __expf(-x));
}

// ---------------------------------------------------------------------------
// Embedding gather: x[b, :] = emb[in_word[b], :]
// ---------------------------------------------------------------------------
__global__ void k_embed(const int* __restrict__ iw, const float* __restrict__ emb,
                        float* __restrict__ x) {
    int b = blockIdx.x;
    int t = threadIdx.x;                       // 128 threads, 4 floats each
    const float4* src = (const float4*)(emb + (size_t)iw[b] * EDIM);
    ((float4*)(x + (size_t)b * EDIM))[t] = src[t];
}

// ---------------------------------------------------------------------------
// Pack u_a [2048,1024] f32 -> bf16 in WMMA-B fragment-native layout.
// One wave per (ktile 32, ntile 16) fragment; lane l: col = nt*16+(l&15),
// K rows kt*32 + (l>>4)*16 + e, e<16  -> 32 contiguous bytes per lane.
// grid = 512 blocks x 256 threads (4096 fragments)
// ---------------------------------------------------------------------------
__global__ __launch_bounds__(256)
void k_pack_w(const float* __restrict__ W, uint4* __restrict__ Wq) {
    const int lane = threadIdx.x & 31;
    const int wid  = blockIdx.x * 8 + (threadIdx.x >> 5);   // 0..4095
    const int kt = wid >> 6;            // 64 K tiles
    const int tn = wid & 63;            // 64 N tiles
    const int col   = tn * 16 + (lane & 15);
    const int kbase = kt * 32 + (lane >> 4) * 16;
    const float* wp = W + (size_t)kbase * HDIM + col;
    FragU f;
    #pragma unroll
    for (int i = 0; i < 8; ++i)
        f.u[i] = pk2(wp[(2 * i) * HDIM], wp[(2 * i + 1) * HDIM]);
    size_t o = ((size_t)wid * 32 + lane) * 2;
    Wq[o] = f.q[0]; Wq[o + 1] = f.q[1];
}

// ---------------------------------------------------------------------------
// Generic small GEMM: C[128,1024] (+)= A[128,K] @ W[K,1024] + bias
// One wave per 16x16 tile, bf16 WMMA, f32 accumulate. K % 32 == 0.
// grid = 64 blocks x 256 threads  (512 waves = 8*64 tiles)
// ---------------------------------------------------------------------------
template <int ACC>
__global__ __launch_bounds__(256)
void k_gemm128(const float* __restrict__ A, const float* __restrict__ W,
               const float* __restrict__ bias, float* __restrict__ C, int K) {
    const int lane = threadIdx.x & 31;
    const int wid  = blockIdx.x * 8 + (threadIdx.x >> 5);
    const int tm = wid & 7;            // M tiles: 128/16 = 8
    const int tn = wid >> 3;           // N tiles: 1024/16 = 64
    const int m0 = tm << 4, n0 = tn << 4;
    const int half = lane >> 4;
    const int l16  = lane & 15;

    v8f acc = {};
    const int rowA = m0 + l16;
    const int colB = n0 + l16;

    for (int k0 = 0; k0 < K; k0 += 32) {
        FragU fa, fb;
        // A fragment: K = k0+16*half .. +15 (contiguous)
        const float* ap = A + (size_t)rowA * K + k0 + half * 16;
        #pragma unroll
        for (int i = 0; i < 4; ++i) fa.u[i]     = pk2(ap[2 * i],      ap[2 * i + 1]);
        #pragma unroll
        for (int i = 0; i < 4; ++i) fa.u[4 + i] = pk2(ap[8 + 2 * i], ap[9 + 2 * i]);
        // B fragment: K rows k0+16*half .. +15 of column colB
        const float* wp = W + (size_t)(k0 + half * 16) * HDIM + colB;
        #pragma unroll
        for (int i = 0; i < 8; ++i)
            fb.u[i] = pk2(wp[(2 * i) * HDIM], wp[(2 * i + 1) * HDIM]);

        acc = __builtin_amdgcn_wmma_f32_16x16x32_bf16(false, fa.v, false, fb.v,
                                                      (short)0, acc, false, false);
    }

    #pragma unroll
    for (int j = 0; j < 8; ++j) {
        int row = m0 + half * 8 + j;
        int col = n0 + l16;
        float v = acc[j] + bias[col];
        float* cp = C + (size_t)row * HDIM + col;
        if (ACC) v += *cp;
        *cp = v;
    }
}

// ---------------------------------------------------------------------------
// init score = v_a.bias, a = 0
// ---------------------------------------------------------------------------
__global__ void k_init(float* __restrict__ score, const float* __restrict__ vab,
                       float* __restrict__ abuf) {
    int i = blockIdx.x * blockDim.x + threadIdx.x;
    if (i < SLEN * BATCH) score[i] = vab[0];
    if (i < BATCH * CHDIM) abuf[i] = 0.0f;
}

// ---------------------------------------------------------------------------
// Fused attention scores:
//   score[t,b] += sum_n v_a[n] * tanh( (enc[t,b,:]@u_a)[n] + b_ua[n] + wa_s[b,n] )
// Workgroup: 32 flat rows (two 16-row WMMA tiles), 8 waves x 8 column tiles
// cover N=1024; each B fragment is shared by both row tiles (halves L2
// weight traffic). B pre-packed bf16 fragment-native; A staged via LDS
// (async DMA double-buffer when available).  grid = 1024 x 256.
// ---------------------------------------------------------------------------
__global__ __launch_bounds__(256)
void k_attn_scores(const float* __restrict__ enc, const uint4* __restrict__ Wq,
                   const float* __restrict__ wa_s, const float* __restrict__ bua,
                   const float* __restrict__ vaw, float* __restrict__ score) {
    const int m0   = blockIdx.x << 5;          // 32 flat rows in [S*B]
    const int tid  = threadIdx.x;
    const int wave = tid >> 5;
    const int lane = tid & 31;
    const int half = lane >> 4;
    const int l16  = lane & 15;
    const int cbase = wave << 7;               // 128 columns per wave

    __shared__ unsigned AsU[32 * 68];          // bf16 tile, padded stride 136

    v8f acc[16];                               // [row-tile g][col-tile i2]
    #pragma unroll
    for (int i = 0; i < 16; ++i) acc[i] = v8f{};

#if USE_ASYNC_LDS
    __shared__ float Af[2][32 * 128];          // f32 double buffer (2 x 16KB)
    auto issue_chunk = [&](int kc, int buf) {
        #pragma unroll
        for (int j = 0; j < 4; ++j) {
            int u   = j * 256 + tid;           // 16-byte unit
            int row = u >> 5, seg = u & 31;
            const float* gp = enc + (size_t)(m0 + row) * CHDIM + kc + seg * 4;
            float* lp = &Af[buf][u * 4];
            __builtin_amdgcn_global_load_async_to_lds_b128(
                (AS1 v4i*)gp, (AS3 v4i*)lp, 0, 0);
        }
    };
    issue_chunk(0, 0);
#endif

    for (int ci = 0; ci < CHDIM / 128; ++ci) {
        const int kc = ci << 7;
#if USE_ASYNC_LDS
        __builtin_amdgcn_s_wait_asynccnt(0);
        __syncthreads();                        // async data visible; As free
        if (ci + 1 < CHDIM / 128) issue_chunk((ci + 1) << 7, (ci + 1) & 1);
        {   // convert staged f32 chunk -> bf16 fragment tile (16 elems/thread)
            int idx = tid * 16;
            int row = idx >> 7, kk = idx & 127;
            const float* s16 = &Af[ci & 1][idx];
            uint4 p0, p1;
            p0.x = pk2(s16[0],  s16[1]);  p0.y = pk2(s16[2],  s16[3]);
            p0.z = pk2(s16[4],  s16[5]);  p0.w = pk2(s16[6],  s16[7]);
            p1.x = pk2(s16[8],  s16[9]);  p1.y = pk2(s16[10], s16[11]);
            p1.z = pk2(s16[12], s16[13]); p1.w = pk2(s16[14], s16[15]);
            uint4* dst = (uint4*)&AsU[row * 68 + (kk >> 1)];
            dst[0] = p0; dst[1] = p1;
        }
        __syncthreads();
#else
        __syncthreads();
        {   // direct cooperative stage: 32 rows x 128 K (f32 -> bf16)
            int idx = tid * 16;
            int row = idx >> 7, kk = idx & 127;
            const float4* src =
                (const float4*)(enc + (size_t)(m0 + row) * CHDIM + kc + kk);
            float4 f0 = src[0], f1 = src[1], f2 = src[2], f3 = src[3];
            uint4 p0, p1;
            p0.x = pk2(f0.x, f0.y); p0.y = pk2(f0.z, f0.w);
            p0.z = pk2(f1.x, f1.y); p0.w = pk2(f1.z, f1.w);
            p1.x = pk2(f2.x, f2.y); p1.y = pk2(f2.z, f2.w);
            p1.z = pk2(f3.x, f3.y); p1.w = pk2(f3.z, f3.w);
            uint4* dst = (uint4*)&AsU[row * 68 + (kk >> 1)];
            dst[0] = p0; dst[1] = p1;
        }
        __syncthreads();
#endif

        #pragma unroll
        for (int ks = 0; ks < 128; ks += 32) {
            FragU fa0, fa1;                    // rows l16 and 16+l16
            fa0.q[0] = *(const uint4*)&AsU[l16 * 68 + ((ks + half * 16) >> 1)];
            fa0.q[1] = *(const uint4*)&AsU[l16 * 68 + ((ks + 8 + half * 16) >> 1)];
            fa1.q[0] = *(const uint4*)&AsU[(16 + l16) * 68 + ((ks + half * 16) >> 1)];
            fa1.q[1] = *(const uint4*)&AsU[(16 + l16) * 68 + ((ks + 8 + half * 16) >> 1)];
            const int ktile = (kc + ks) >> 5;
            #pragma unroll
            for (int i2 = 0; i2 < 8; ++i2) {
                FragU fb;
                const uint4* bp =
                    Wq + ((size_t)(ktile * 64 + wave * 8 + i2) * 32 + lane) * 2;
                fb.q[0] = bp[0]; fb.q[1] = bp[1];
                acc[i2] = __builtin_amdgcn_wmma_f32_16x16x32_bf16(
                    false, fa0.v, false, fb.v, (short)0, acc[i2], false, false);
                acc[8 + i2] = __builtin_amdgcn_wmma_f32_16x16x32_bf16(
                    false, fa1.v, false, fb.v, (short)0, acc[8 + i2], false, false);
            }
        }
    }

    // epilogue: tanh + dot with v_a, fused; two 16-row groups
    float p[2][8];
    #pragma unroll
    for (int g = 0; g < 2; ++g)
        #pragma unroll
        for (int j = 0; j < 8; ++j) p[g][j] = 0.0f;
    const int b0 = m0 & (BATCH - 1);           // 32-row tile spans a single t
    #pragma unroll
    for (int i2 = 0; i2 < 8; ++i2) {
        int col = cbase + (i2 << 4) + l16;
        float va = vaw[col], bb = bua[col];
        #pragma unroll
        for (int g = 0; g < 2; ++g) {
            #pragma unroll
            for (int j = 0; j < 8; ++j) {
                int b = b0 + g * 16 + half * 8 + j;
                float v = acc[g * 8 + i2][j] + wa_s[(size_t)b * HDIM + col] + bb;
                p[g][j] += tanhf(v) * va;
            }
        }
    }
    #pragma unroll
    for (int g = 0; g < 2; ++g) {
        #pragma unroll
        for (int j = 0; j < 8; ++j) {
            float v = p[g][j];
            v += __shfl_xor(v, 1, 32);
            v += __shfl_xor(v, 2, 32);
            v += __shfl_xor(v, 4, 32);
            v += __shfl_xor(v, 8, 32);
            p[g][j] = v;
        }
    }
    if (l16 == 0) {
        #pragma unroll
        for (int g = 0; g < 2; ++g)
            #pragma unroll
            for (int j = 0; j < 8; ++j)
                atomicAdd(&score[m0 + g * 16 + half * 8 + j], p[g][j]);
    }
}

// ---------------------------------------------------------------------------
// Softmax over S per batch column. grid = 128 blocks x 256 threads (t = tid)
// ---------------------------------------------------------------------------
__global__ __launch_bounds__(256)
void k_softmax(const float* __restrict__ score, float* __restrict__ probs) {
    int b = blockIdx.x, t = threadIdx.x;
    __shared__ float red[256];
    float v = score[t * BATCH + b];
    red[t] = v; __syncthreads();
    for (int off = 128; off > 0; off >>= 1) {
        if (t < off) red[t] = fmaxf(red[t], red[t + off]);
        __syncthreads();
    }
    float mx = red[0]; __syncthreads();
    float e = __expf(v - mx);
    red[t] = e; __syncthreads();
    for (int off = 128; off > 0; off >>= 1) {
        if (t < off) red[t] += red[t + off];
        __syncthreads();
    }
    probs[t * BATCH + b] = e / red[0];
}

// ---------------------------------------------------------------------------
// Attended state: a[b,:] += sum_{t in group} probs[t,b] * enc[t,b,:]
// grid = (128, 8), 256 threads; each thread owns 8 channels.
// ---------------------------------------------------------------------------
__global__ __launch_bounds__(256)
void k_attend(const float* __restrict__ probs, const float* __restrict__ enc,
              float* __restrict__ abuf) {
    int b = blockIdx.x, g = blockIdx.y;
    int c = threadIdx.x * 8;
    float s0 = 0, s1 = 0, s2 = 0, s3 = 0, s4 = 0, s5 = 0, s6 = 0, s7 = 0;
    for (int t = g * 32; t < g * 32 + 32; ++t) {
        float pv = probs[t * BATCH + b];
        const float4* e4 =
            (const float4*)(enc + ((size_t)t * BATCH + b) * CHDIM + c);
        float4 v0 = e4[0], v1 = e4[1];
        s0 += pv * v0.x; s1 += pv * v0.y; s2 += pv * v0.z; s3 += pv * v0.w;
        s4 += pv * v1.x; s5 += pv * v1.y; s6 += pv * v1.z; s7 += pv * v1.w;
    }
    float* ap = abuf + (size_t)b * CHDIM + c;
    atomicAdd(ap + 0, s0); atomicAdd(ap + 1, s1);
    atomicAdd(ap + 2, s2); atomicAdd(ap + 3, s3);
    atomicAdd(ap + 4, s4); atomicAdd(ap + 5, s5);
    atomicAdd(ap + 6, s6); atomicAdd(ap + 7, s7);
}

// ---------------------------------------------------------------------------
// Gates: z = sigmoid(zpre) (in place), r = sigmoid(rpre) (in place), rs = r*s
// ---------------------------------------------------------------------------
__global__ void k_gates(float* __restrict__ zpre, float* __restrict__ rpre,
                        const float* __restrict__ s, float* __restrict__ rs) {
    int i = blockIdx.x * blockDim.x + threadIdx.x;
    if (i >= BATCH * HDIM) return;
    float z = sigmoidf_(zpre[i]);
    float r = sigmoidf_(rpre[i]);
    zpre[i] = z;
    rpre[i] = r;
    rs[i] = r * s[i];
}

// s_tilda = tanh(spre); s_new = (1-z)*s + z*s_tilda
__global__ void k_newstate(const float* __restrict__ spre, const float* __restrict__ z,
                           const float* __restrict__ s, float* __restrict__ snew) {
    int i = blockIdx.x * blockDim.x + threadIdx.x;
    if (i >= BATCH * HDIM) return;
    float st = tanhf(spre[i]);
    float zz = z[i];
    snew[i] = (1.0f - zz) * s[i] + zz * st;
}

__global__ void k_relu(const float* __restrict__ in, float* __restrict__ out) {
    int i = blockIdx.x * blockDim.x + threadIdx.x;
    if (i >= BATCH * HDIM) return;
    out[i] = fmaxf(in[i], 0.0f);
}

// ---------------------------------------------------------------------------
extern "C" void kernel_launch(void* const* d_in, const int* in_sizes, int n_in,
                              void* d_out, int out_size, void* d_ws, size_t ws_size,
                              hipStream_t stream) {
    (void)in_sizes; (void)n_in; (void)out_size; (void)ws_size;

    const int*   in_word = (const int*)  d_in[0];
    const float* enc     = (const float*)d_in[1];
    const float* s       = (const float*)d_in[2];
    const float* emb     = (const float*)d_in[3];
    const float* wa_w = (const float*)d_in[4],  *wa_b = (const float*)d_in[5];
    const float* ua_w = (const float*)d_in[6],  *ua_b = (const float*)d_in[7];
    const float* va_w = (const float*)d_in[8],  *va_b = (const float*)d_in[9];
    const float* w_w  = (const float*)d_in[10], *w_b  = (const float*)d_in[11];
    const float* wz_w = (const float*)d_in[12], *wz_b = (const float*)d_in[13];
    const float* wr_w = (const float*)d_in[14], *wr_b = (const float*)d_in[15];
    const float* u_w  = (const float*)d_in[16], *u_b  = (const float*)d_in[17];
    const float* uz_w = (const float*)d_in[18], *uz_b = (const float*)d_in[19];
    const float* ur_w = (const float*)d_in[20], *ur_b = (const float*)d_in[21];
    const float* c_w  = (const float*)d_in[22], *c_b  = (const float*)d_in[23];
    const float* cz_w = (const float*)d_in[24], *cz_b = (const float*)d_in[25];
    const float* cr_w = (const float*)d_in[26], *cr_b = (const float*)d_in[27];
    const float* uo_w = (const float*)d_in[28], *uo_b = (const float*)d_in[29];
    const float* vo_w = (const float*)d_in[30], *vo_b = (const float*)d_in[31];
    const float* co_w = (const float*)d_in[32], *co_b = (const float*)d_in[33];

    float* out   = (float*)d_out;
    float* snew  = out;                        // [128,1024]
    float* t_i   = out + BATCH * HDIM;         // [128,1024]
    float* probs = out + 2 * BATCH * HDIM;     // [256,128]

    float* ws = (float*)d_ws;
    float* x     = ws;                          size_t o = BATCH * EDIM;
    float* wa_s  = ws + o; o += BATCH * HDIM;
    float* score = ws + o; o += SLEN * BATCH;
    float* abuf  = ws + o; o += BATCH * CHDIM;
    float* zpre  = ws + o; o += BATCH * HDIM;
    float* rpre  = ws + o; o += BATCH * HDIM;
    float* spre  = ws + o; o += BATCH * HDIM;
    float* opre  = ws + o; o += BATCH * HDIM;
    float* rs    = ws + o; o += BATCH * HDIM;
    uint4* uaPk  = (uint4*)(ws + o); o += (size_t)CHDIM * HDIM / 2;  // bf16 packed

    const dim3 tb(256);
    const int ebN = (BATCH * HDIM) / 256;       // 512 blocks for elementwise

    // 1. embedding gather + weight pre-pack (independent)
    k_embed<<<BATCH, 128, 0, stream>>>(in_word, emb, x);
    k_pack_w<<<512, tb, 0, stream>>>(ua_w, uaPk);
    // 2. wa_s = s @ w_a + b
    k_gemm128<0><<<64, tb, 0, stream>>>(s, wa_w, wa_b, wa_s, HDIM);
    // 3. init score (= v_a bias) and attended accumulator
    k_init<<<1024, tb, 0, stream>>>(score, va_b, abuf);
    // 4. fused big GEMM + tanh + v_a dot (32-row M-tiles)
    k_attn_scores<<<(SLEN * BATCH) / 32, tb, 0, stream>>>(enc, uaPk, wa_s, ua_b,
                                                          va_w, score);
    // 5. softmax over source positions
    k_softmax<<<BATCH, tb, 0, stream>>>(score, probs);
    // 6. attended state a
    k_attend<<<dim3(BATCH, 8), tb, 0, stream>>>(probs, enc, abuf);

    // 7. gate pre-activations
    k_gemm128<0><<<64, tb, 0, stream>>>(x,    wz_w, wz_b, zpre, EDIM);
    k_gemm128<1><<<64, tb, 0, stream>>>(s,    uz_w, uz_b, zpre, HDIM);
    k_gemm128<1><<<64, tb, 0, stream>>>(abuf, cz_w, cz_b, zpre, CHDIM);

    k_gemm128<0><<<64, tb, 0, stream>>>(x,    wr_w, wr_b, rpre, EDIM);
    k_gemm128<1><<<64, tb, 0, stream>>>(s,    ur_w, ur_b, rpre, HDIM);
    k_gemm128<1><<<64, tb, 0, stream>>>(abuf, cr_w, cr_b, rpre, CHDIM);

    k_gemm128<0><<<64, tb, 0, stream>>>(x,    w_w,  w_b,  spre, EDIM);
    k_gemm128<1><<<64, tb, 0, stream>>>(abuf, c_w,  c_b,  spre, CHDIM);

    // 8. z, r, r*s
    k_gates<<<ebN, tb, 0, stream>>>(zpre, rpre, s, rs);
    // 9. spre += (r*s) @ u + b
    k_gemm128<1><<<64, tb, 0, stream>>>(rs, u_w, u_b, spre, HDIM);
    // 10. s_new
    k_newstate<<<ebN, tb, 0, stream>>>(spre, zpre, s, snew);
    // 11. output pre-activation
    k_gemm128<0><<<64, tb, 0, stream>>>(x,    vo_w, vo_b, opre, EDIM);
    k_gemm128<1><<<64, tb, 0, stream>>>(abuf, co_w, co_b, opre, CHDIM);
    k_gemm128<1><<<64, tb, 0, stream>>>(snew, uo_w, uo_b, opre, HDIM);
    // 12. t_i = relu(opre)
    k_relu<<<ebN, tb, 0, stream>>>(opre, t_i);
}
